// LSTMClassifier_64398739636765
// MI455X (gfx1250) — matrix operand order repaired and checked by reading, not compiled, not run on hardware
//
#include <hip/hip_runtime.h>
#include <cstddef>

// Problem constants (match reference)
#define B_   64
#define T_   512
#define DIN_ 256
#define H_   512
#define G4_  (4 * H_)   // 2048 gate rows

typedef __attribute__((ext_vector_type(16))) __bf16 v16bf;
typedef __attribute__((ext_vector_type(8)))  float  v8f;

union FragU { uint4 q[2]; v16bf v; };

// ---- bf16 helpers (RNE) ----
__device__ __forceinline__ unsigned short f2bf(float f) {
  unsigned u = __float_as_uint(f);
  unsigned r = u + 0x7FFFu + ((u >> 16) & 1u);
  return (unsigned short)(r >> 16);
}
__device__ __forceinline__ float bf2f(unsigned short h) {
  return __uint_as_float(((unsigned)h) << 16);
}
__device__ __forceinline__ float sigmoidf_(float x) {
  return 1.0f / (1.0f + __expf(-x));
}

// ---- WMMA fragment loaders (CDNA5 16-bit layouts, wave32) ----
// A (16x32, MxK): lane&15 = M row; elements 0..7  = K[koff .. koff+7],
//                 elements 8..15 = K[koff+16 .. koff+23], koff = kbase + (lane>>4)*8
__device__ __forceinline__ v16bf load_fragA(const unsigned short* row, int koff) {
  FragU f;
  f.q[0] = *reinterpret_cast<const uint4*>(row + koff);
  f.q[1] = *reinterpret_cast<const uint4*>(row + koff + 16);
  return f.v;
}
// B (32x16, KxN): lane&15 = N col; elements 0..7 = K[koff .. koff+7],
// elements 8..15 = K[koff+16 .. koff+23], koff = kbase + (lane>>4)*8 (row-major [N][K] weight row)
__device__ __forceinline__ v16bf load_fragB(const unsigned short* row, int koff) {
  FragU f;
  f.q[0] = *reinterpret_cast<const uint4*>(row + koff);
  f.q[1] = *reinterpret_cast<const uint4*>(row + koff + 16);
  return f.v;
}

// ---- cross-workgroup phase barrier (16 WGs per batch-tile group) ----
__device__ __forceinline__ void group_barrier(unsigned int* bar, unsigned int members,
                                              unsigned int& phase) {
  __threadfence();
  __syncthreads();
  if (threadIdx.x == 0) {
    ++phase;
    __hip_atomic_fetch_add(bar, 1u, __ATOMIC_RELEASE, __HIP_MEMORY_SCOPE_AGENT);
    while (__hip_atomic_load(bar, __ATOMIC_ACQUIRE, __HIP_MEMORY_SCOPE_AGENT) <
           members * phase) {
      __builtin_amdgcn_s_sleep(8);
    }
  }
  __syncthreads();
  __threadfence();
}

// ---- small utility kernels ----
__global__ void convert_f2bf(const float* __restrict__ s, unsigned short* __restrict__ d, int n) {
  for (int i = blockIdx.x * blockDim.x + threadIdx.x; i < n; i += gridDim.x * blockDim.x)
    d[i] = f2bf(s[i]);
}
__global__ void add_bias(const float* __restrict__ a, const float* __restrict__ b,
                         float* __restrict__ d, int n) {
  for (int i = blockIdx.x * blockDim.x + threadIdx.x; i < n; i += gridDim.x * blockDim.x)
    d[i] = a[i] + b[i];
}
__global__ void zero_cnt(unsigned int* c, int n) {
  int i = blockIdx.x * blockDim.x + threadIdx.x;
  if (i < n) c[i] = 0u;
}

// ---- persistent 2-layer LSTM recurrence ----
// Grid: 64 WGs = 4 batch-tile groups x 16 hidden chunks. Block: 256 threads = 8 waves.
// Wave w: gate type gt = w>>1, column half hf = w&1 -> one 16x16 WMMA output tile.
__global__ __launch_bounds__(256) void lstm_persistent(
    const unsigned short* __restrict__ xb,
    const unsigned short* __restrict__ wih0, const unsigned short* __restrict__ whh0,
    const float* __restrict__ bias0,
    const unsigned short* __restrict__ wih1, const unsigned short* __restrict__ whh1,
    const float* __restrict__ bias1,
    unsigned short* __restrict__ h0seq,   // [B][T][H] bf16 (layer-0 output sequence)
    unsigned short* __restrict__ hbuf,    // [2][B][H] bf16 ping-pong h state
    unsigned int* __restrict__ cnt)       // [4] group barrier counters
{
  __shared__ float gates[4 * 16 * 32];    // [gate][batch_local][hidden_local]
  __shared__ float cst[16 * 32];          // cell state for this WG's chunk

  const int tid  = threadIdx.x;
  const int lane = tid & 31;
  const int wave = tid >> 5;
  const int gt   = wave >> 1;             // 0..3 : i,f,g,o
  const int hf   = wave & 1;              // column half within 32-col chunk
  const int hi   = lane >> 4;
  const int l15  = lane & 15;
  const int grp   = blockIdx.x >> 4;      // batch tile (0..3)
  const int chunk = blockIdx.x & 15;      // hidden chunk (0..15)
  const int c0    = chunk * 32;

  unsigned int  phase = 0;
  unsigned int* bar   = cnt + grp;

  const int b = grp * 16 + l15;                   // A-fragment batch row for this lane
  const int n = gt * H_ + c0 + hf * 16 + l15;     // B-fragment gate row for this lane

  for (int layer = 0; layer < 2; ++layer) {
    const unsigned short* aseq = layer ? h0seq : xb;
    const int din              = layer ? H_ : DIN_;
    const unsigned short* wih  = layer ? wih1 : wih0;
    const unsigned short* whh  = layer ? whh1 : whh0;
    const float* bias          = layer ? bias1 : bias0;
    unsigned short* hseq_out   = layer ? nullptr : h0seq;

    // zero cell state + this WG's region of both h ping-pong buffers
    for (int item = tid; item < 16 * 32; item += 256) {
      cst[item] = 0.0f;
      int bl = item >> 5, j = item & 31;
      size_t col = (size_t)(grp * 16 + bl) * H_ + c0 + j;
      hbuf[col]                    = 0;
      hbuf[(size_t)B_ * H_ + col]  = 0;
    }
    group_barrier(bar, 16, phase);

    const unsigned short* browI = wih + (size_t)n * din;
    const unsigned short* browH = whh + (size_t)n * H_;
    const float biasn = bias[n];

    for (int t = 0; t < T_; ++t) {
      const unsigned short* arow = aseq + ((size_t)b * T_ + t) * din;
      const unsigned short* hrow = hbuf + (size_t)(t & 1) * B_ * H_ + (size_t)b * H_;

      v8f acc;
      #pragma unroll
      for (int i = 0; i < 8; ++i) acc[i] = biasn;

      // gates += x_t (or h0_t) @ w_ih^T   (K = din)
      #pragma unroll 4
      for (int k = 0; k < din; k += 32)
        acc = __builtin_amdgcn_wmma_f32_16x16x32_bf16(
            false, load_fragA(arow, k + hi * 8),
            false, load_fragB(browI, k + hi * 8),
            (short)0, acc, false, false);

      // gates += h_{t-1} @ w_hh^T         (K = H)
      #pragma unroll 4
      for (int k = 0; k < H_; k += 32)
        acc = __builtin_amdgcn_wmma_f32_16x16x32_bf16(
            false, load_fragA(hrow, k + hi * 8),
            false, load_fragB(browH, k + hi * 8),
            (short)0, acc, false, false);

      // scatter 16x16 f32 tile to LDS: D layout: VGPR i -> M = i + 8*(lane>>4), N = lane&15
      #pragma unroll
      for (int i = 0; i < 8; ++i)
        gates[(gt * 16 + i + 8 * hi) * 32 + hf * 16 + l15] = acc[i];
      __syncthreads();

      // elementwise LSTM cell for this WG's [16 batch x 32 hidden] chunk
      unsigned short* hout = hbuf + (size_t)((t + 1) & 1) * B_ * H_;
      for (int item = tid; item < 512; item += 256) {
        int bl = item >> 5, j = item & 31;
        float ig = sigmoidf_(gates[(0 * 16 + bl) * 32 + j]);
        float fg = sigmoidf_(gates[(1 * 16 + bl) * 32 + j]);
        float gg = tanhf   (gates[(2 * 16 + bl) * 32 + j]);
        float og = sigmoidf_(gates[(3 * 16 + bl) * 32 + j]);
        float c  = fg * cst[item] + ig * gg;
        cst[item] = c;
        unsigned short hb = f2bf(og * tanhf(c));
        size_t gcol = (size_t)(grp * 16 + bl) * H_ + c0 + j;
        hout[gcol] = hb;
        if (hseq_out)
          hseq_out[((size_t)(grp * 16 + bl) * T_ + t) * H_ + c0 + j] = hb;
      }
      group_barrier(bar, 16, phase);   // h_t fully published before step t+1 reads it
    }
  }
}

// ---- final FC + sigmoid: out[b] = sigmoid(h_last[b,:] . w_fc + b_fc) ----
__global__ void fc_sigmoid(const unsigned short* __restrict__ hfin,
                           const float* __restrict__ wfc, const float* __restrict__ bfc,
                           float* __restrict__ out) {
  __shared__ float red[64];
  int bb = blockIdx.x;
  float s = 0.0f;
  for (int j = threadIdx.x; j < H_; j += 64)
    s += bf2f(hfin[(size_t)bb * H_ + j]) * wfc[j];
  red[threadIdx.x] = s;
  __syncthreads();
  for (int off = 32; off > 0; off >>= 1) {
    if ((int)threadIdx.x < off) red[threadIdx.x] += red[threadIdx.x + off];
    __syncthreads();
  }
  if (threadIdx.x == 0) out[bb] = 1.0f / (1.0f + __expf(-(red[0] + bfc[0])));
}

extern "C" void kernel_launch(void* const* d_in, const int* in_sizes, int n_in,
                              void* d_out, int out_size, void* d_ws, size_t ws_size,
                              hipStream_t stream) {
  (void)in_sizes; (void)n_in; (void)out_size; (void)ws_size;
  const float* x     = (const float*)d_in[0];
  const float* w_ih0 = (const float*)d_in[1];
  const float* w_hh0 = (const float*)d_in[2];
  const float* b_ih0 = (const float*)d_in[3];
  const float* b_hh0 = (const float*)d_in[4];
  const float* w_ih1 = (const float*)d_in[5];
  const float* w_hh1 = (const float*)d_in[6];
  const float* b_ih1 = (const float*)d_in[7];
  const float* b_hh1 = (const float*)d_in[8];
  const float* w_fc  = (const float*)d_in[9];
  const float* b_fc  = (const float*)d_in[10];
  float* out = (float*)d_out;

  // Workspace carve (256B aligned): ~58 MB total
  char* p = (char*)d_ws;
  auto carve = [&](size_t bytes) -> void* {
    void* r = (void*)p;
    p += (bytes + 255) & ~(size_t)255;
    return r;
  };
  unsigned short* xb    = (unsigned short*)carve((size_t)B_ * T_ * DIN_ * 2);
  unsigned short* wih0b = (unsigned short*)carve((size_t)G4_ * DIN_ * 2);
  unsigned short* whh0b = (unsigned short*)carve((size_t)G4_ * H_ * 2);
  unsigned short* wih1b = (unsigned short*)carve((size_t)G4_ * H_ * 2);
  unsigned short* whh1b = (unsigned short*)carve((size_t)G4_ * H_ * 2);
  float*          bias0 = (float*)carve((size_t)G4_ * 4);
  float*          bias1 = (float*)carve((size_t)G4_ * 4);
  unsigned short* h0seq = (unsigned short*)carve((size_t)B_ * T_ * H_ * 2);
  unsigned short* hbuf  = (unsigned short*)carve((size_t)2 * B_ * H_ * 2);
  unsigned int*   cnt   = (unsigned int*)carve(64);

  convert_f2bf<<<256, 256, 0, stream>>>(x, xb, B_ * T_ * DIN_);
  convert_f2bf<<<64,  256, 0, stream>>>(w_ih0, wih0b, G4_ * DIN_);
  convert_f2bf<<<128, 256, 0, stream>>>(w_hh0, whh0b, G4_ * H_);
  convert_f2bf<<<128, 256, 0, stream>>>(w_ih1, wih1b, G4_ * H_);
  convert_f2bf<<<128, 256, 0, stream>>>(w_hh1, whh1b, G4_ * H_);
  add_bias<<<8, 256, 0, stream>>>(b_ih0, b_hh0, bias0, G4_);
  add_bias<<<8, 256, 0, stream>>>(b_ih1, b_hh1, bias1, G4_);
  zero_cnt<<<1, 64, 0, stream>>>(cnt, 16);

  lstm_persistent<<<64, 256, 0, stream>>>(xb, wih0b, whh0b, bias0,
                                          wih1b, whh1b, bias1,
                                          h0seq, hbuf, cnt);

  // h_511 lands in ping-pong buffer 0 (t=511 writes buf[(511+1)&1] = buf[0])
  fc_sigmoid<<<B_, 64, 0, stream>>>(hbuf, w_fc, b_fc, out);
}
